// Seq2Seq_56659208569221
// MI455X (gfx1250) — matrix operand order repaired and checked by reading, not compiled, not run on hardware
//
#include <hip/hip_runtime.h>

typedef __attribute__((ext_vector_type(16))) _Float16 v16h;
typedef __attribute__((ext_vector_type(8)))  float    v8f;

#define V_   8000
#define E_   512
#define H_   1024
#define B_   64
#define S_   128
#define T_   128
#define G4H  (4 * H_)

// ---------------------------------------------------------------------------
// A tile 16x32 f16, row-major in LDS with stride 32 (CDNA5 ISA 7.12.2 layout).
// lane: m = lane&15, half = lane>>4 ; element e: k = 16*(e>>3) + 8*half + (e&7)
// Per-lane halves contiguous in groups of 8 -> 2x ds_load_b128 per tile.
// ---------------------------------------------------------------------------
static __device__ __forceinline__ v16h load_a_lds(const _Float16* __restrict__ As, int lane) {
  v16h a;
  const int m  = lane & 15;
  const int h8 = (lane >> 4) << 3;
#pragma unroll
  for (int e = 0; e < 16; ++e) {
    const int k = ((e >> 3) << 4) + h8 + (e & 7);
    a[e] = As[m * 32 + k];
  }
  return a;
}

static __device__ __forceinline__ float sigf(float x) { return 1.0f / (1.0f + expf(-x)); }

// ---------------------------------------------------------------------------
// Shared WMMA mainloop body (macro-free via inline function is awkward with
// LDS, so both GEMM kernels repeat the small pipelined loop).
// Wb layout (k_pack_b): element (k,n') stored at
//   Wb[(((k/32)*nTilesTot + n'/16)*32 + lane(k,n'))*16 + e(k,n')]
// so each lane reads its 16-half B fragment as one contiguous 32B chunk.
// ---------------------------------------------------------------------------

// Generic GEMM: out[M x N](f32) = concat(a0,a1,a2)[M x K](f16) * Wb + bias
// M = 64 * gridDim.y. Block = 256 threads = 8 waves; wave owns a 64x16 strip.
template <bool GUARD>
__global__ void k_gemm(const _Float16* __restrict__ a0, int lda0, int K0,
                       const _Float16* __restrict__ a1, int lda1, int K1,
                       const _Float16* __restrict__ a2, int lda2, int K2,
                       const _Float16* __restrict__ Wb, int nTilesTot,
                       float* __restrict__ out, long long ldOut,
                       const float* __restrict__ bias)
{
  __shared__ _Float16 As[2][64 * 32];
  const int lane  = threadIdx.x & 31;
  const int wave  = threadIdx.x >> 5;
  const int nTile = blockIdx.x * 8 + wave;
  const int mBase = blockIdx.y * 64;
  // scalar (SGPR) predicate so WMMA never sits under an EXEC mask
  const bool valid = !GUARD || (__builtin_amdgcn_readfirstlane(nTile) < nTilesTot);
  const int K  = K0 + K1 + K2;
  const int KP = K >> 5;

  const int rA = threadIdx.x >> 2;
  const int cA = (threadIdx.x & 3) << 3;

  auto loadA = [&](int kp) -> uint4 {
    int kk = (kp << 5) + cA;
    const _Float16* src; int ld;
    if (kk < K0)           { src = a0; ld = lda0; }
    else if (kk < K0 + K1) { src = a1; ld = lda1; kk -= K0; }
    else                   { src = a2; ld = lda2; kk -= K0 + K1; }
    return *reinterpret_cast<const uint4*>(src + (size_t)(mBase + rA) * ld + kk);
  };

  v8f acc[4] = {};
  uint4 stage = loadA(0);
  *reinterpret_cast<uint4*>(&As[0][rA * 32 + cA]) = stage;
  __syncthreads();

  for (int kp = 0; kp < KP; ++kp) {
    const int  buf  = kp & 1;
    const bool more = (kp + 1) < KP;
    if (more) stage = loadA(kp + 1);

    if (valid) {
      const _Float16* bp = Wb + (((size_t)kp * nTilesTot + nTile) * 32 + lane) * 16;
      __builtin_prefetch(bp + (size_t)nTilesTot * 512, 0, 1);  // next B panel
      const v16h bt  = *reinterpret_cast<const v16h*>(bp);
      // hoist all four A fragments: 8x ds_load_b128, one wait, 4 WMMA b2b
      const v16h at0 = load_a_lds(&As[buf][0 * 512], lane);
      const v16h at1 = load_a_lds(&As[buf][1 * 512], lane);
      const v16h at2 = load_a_lds(&As[buf][2 * 512], lane);
      const v16h at3 = load_a_lds(&As[buf][3 * 512], lane);
      acc[0] = __builtin_amdgcn_wmma_f32_16x16x32_f16(false, at0, false, bt, (short)0, acc[0], false, false);
      acc[1] = __builtin_amdgcn_wmma_f32_16x16x32_f16(false, at1, false, bt, (short)0, acc[1], false, false);
      acc[2] = __builtin_amdgcn_wmma_f32_16x16x32_f16(false, at2, false, bt, (short)0, acc[2], false, false);
      acc[3] = __builtin_amdgcn_wmma_f32_16x16x32_f16(false, at3, false, bt, (short)0, acc[3], false, false);
    }
    if (more) *reinterpret_cast<uint4*>(&As[1 - buf][rA * 32 + cA]) = stage;
    __syncthreads();
  }

  if (valid) {
    // C layout: element (m,n): lane = (n&15) + 16*((m&15)>>3), vgpr = m&7
    const int   nloc = lane & 15;
    const int   mhi  = (lane >> 4) << 3;
    const int   n    = nTile * 16 + nloc;
    const float bv   = bias ? bias[n] : 0.0f;
#pragma unroll
    for (int mt = 0; mt < 4; ++mt) {
#pragma unroll
      for (int r = 0; r < 8; ++r) {
        const int m = mBase + mt * 16 + mhi + r;
        out[(size_t)m * ldOut + n] = acc[mt][r] + bv;
      }
    }
  }
}

// ---------------------------------------------------------------------------
// Fused gates-GEMM + LSTM cell.  N = 4096 gate-interleaved columns:
//   n' = (j/32)*128 + gate*32 + (j%32)   (gate in {i,f,g,o}, j in [0,H))
// so block nb owns ALL four gates for j in [nb*32, nb*32+32).  The biased
// 64x128 gate block is staged in LDS, then c/h updated in-place.
// ---------------------------------------------------------------------------
__global__ void k_gemm_lstm(const _Float16* __restrict__ a0, int lda0, int K0,
                            const _Float16* __restrict__ a1, int lda1, int K1,
                            const _Float16* __restrict__ a2, int lda2, int K2,
                            const _Float16* __restrict__ Wb,
                            const float* __restrict__ bias,
                            float* __restrict__ c,
                            _Float16* __restrict__ h16,
                            _Float16* __restrict__ enc_slot)
{
  __shared__ _Float16 As[2][64 * 32];
  __shared__ float    Cs[64 * 128];
  const int lane  = threadIdx.x & 31;
  const int wave  = threadIdx.x >> 5;
  const int nTile = blockIdx.x * 8 + wave;
  const int nTilesTot = G4H / 16;  // 256, exact: no guard
  const int K  = K0 + K1 + K2;
  const int KP = K >> 5;

  const int rA = threadIdx.x >> 2;
  const int cA = (threadIdx.x & 3) << 3;

  auto loadA = [&](int kp) -> uint4 {
    int kk = (kp << 5) + cA;
    const _Float16* src; int ld;
    if (kk < K0)           { src = a0; ld = lda0; }
    else if (kk < K0 + K1) { src = a1; ld = lda1; kk -= K0; }
    else                   { src = a2; ld = lda2; kk -= K0 + K1; }
    return *reinterpret_cast<const uint4*>(src + (size_t)rA * ld + kk);
  };

  v8f acc[4] = {};
  uint4 stage = loadA(0);
  *reinterpret_cast<uint4*>(&As[0][rA * 32 + cA]) = stage;
  __syncthreads();

  for (int kp = 0; kp < KP; ++kp) {
    const int  buf  = kp & 1;
    const bool more = (kp + 1) < KP;
    if (more) stage = loadA(kp + 1);

    const _Float16* bp = Wb + (((size_t)kp * nTilesTot + nTile) * 32 + lane) * 16;
    __builtin_prefetch(bp + (size_t)nTilesTot * 512, 0, 1);
    const v16h bt  = *reinterpret_cast<const v16h*>(bp);
    const v16h at0 = load_a_lds(&As[buf][0 * 512], lane);
    const v16h at1 = load_a_lds(&As[buf][1 * 512], lane);
    const v16h at2 = load_a_lds(&As[buf][2 * 512], lane);
    const v16h at3 = load_a_lds(&As[buf][3 * 512], lane);
    acc[0] = __builtin_amdgcn_wmma_f32_16x16x32_f16(false, at0, false, bt, (short)0, acc[0], false, false);
    acc[1] = __builtin_amdgcn_wmma_f32_16x16x32_f16(false, at1, false, bt, (short)0, acc[1], false, false);
    acc[2] = __builtin_amdgcn_wmma_f32_16x16x32_f16(false, at2, false, bt, (short)0, acc[2], false, false);
    acc[3] = __builtin_amdgcn_wmma_f32_16x16x32_f16(false, at3, false, bt, (short)0, acc[3], false, false);

    if (more) *reinterpret_cast<uint4*>(&As[1 - buf][rA * 32 + cA]) = stage;
    __syncthreads();
  }

  // stage biased gates (64 x 128 block-local columns) into LDS
  {
    const int   nloc = lane & 15;
    const int   mhi  = (lane >> 4) << 3;
    const int   npl  = wave * 16 + nloc;        // local n' in [0,128)
    const int   gate = npl >> 5;
    const int   jl   = npl & 31;
    const int   n    = gate * H_ + blockIdx.x * 32 + jl;  // original column
    const float bv   = bias[n];
#pragma unroll
    for (int mt = 0; mt < 4; ++mt) {
#pragma unroll
      for (int r = 0; r < 8; ++r) {
        const int m = mt * 16 + mhi + r;
        Cs[m * 128 + npl] = acc[mt][r] + bv;
      }
    }
  }
  __syncthreads();

  // LSTM cell for this block's j slice: j in [blockIdx.x*32, +32), all batches
  for (int idx = threadIdx.x; idx < 64 * 32; idx += 256) {
    const int m  = idx >> 5;
    const int jl = idx & 31;
    const float gi = Cs[m * 128 + jl];
    const float gf = Cs[m * 128 + 32 + jl];
    const float gg = Cs[m * 128 + 64 + jl];
    const float go = Cs[m * 128 + 96 + jl];
    const int   ci = m * H_ + blockIdx.x * 32 + jl;
    const float cn = sigf(gf) * c[ci] + sigf(gi) * tanhf(gg);
    const float hn = sigf(go) * tanhf(cn);
    c[ci]   = cn;
    h16[ci] = (_Float16)hn;
    if (enc_slot) enc_slot[ci] = (_Float16)hn;
  }
}

// ---------------------------------------------------------------------------
// Pack weight W [N x K] f32 into f16 WMMA B-fragment order (optionally with
// the gate-interleave column permutation used by k_gemm_lstm).
// ---------------------------------------------------------------------------
__global__ void k_pack_b(const float* __restrict__ W, _Float16* __restrict__ Wb,
                         int N, int Kpart, int panelStart, int gatePerm)
{
  const int    nTiles = N >> 4;
  const size_t total  = (size_t)N * Kpart;
  _Float16* dst = Wb + (size_t)panelStart * N * 32;
  for (size_t o = (size_t)blockIdx.x * blockDim.x + threadIdx.x; o < total;
       o += (size_t)gridDim.x * blockDim.x) {
    const int    e      = (int)(o & 15);
    const int    L      = (int)((o >> 4) & 31);
    const size_t tIdx   = o >> 9;
    const int    nt     = (int)(tIdx % nTiles);
    const int    kp     = (int)(tIdx / nTiles);
    const int    k      = (kp << 5) + ((e >> 3) << 4) + ((L >> 4) << 3) + (e & 7);
    const int    nPrime = (nt << 4) + (L & 15);
    int n;
    if (gatePerm) {  // n' = (j/32)*128 + gate*32 + (j%32), N == 4096
      const int rem = nPrime & 127;
      n = (rem >> 5) * H_ + ((nPrime >> 7) << 5) + (rem & 31);
    } else {
      n = nPrime;
    }
    dst[o] = (_Float16)W[(size_t)n * Kpart + k];
  }
}

// ---------------------------------------------------------------------------
// Bahdanau attention: one block per batch row.
// ---------------------------------------------------------------------------
__global__ void k_attn(const float* __restrict__ q,
                       const float* __restrict__ enc_proj,
                       const _Float16* __restrict__ enc_out,
                       const float* __restrict__ vvec,
                       _Float16* __restrict__ ctx16)
{
  __shared__ float es[S_];
  __shared__ float ws[S_];
  const int b    = blockIdx.x;
  const int lane = threadIdx.x & 31;
  const int wave = threadIdx.x >> 5;

  for (int s = wave * 16; s < wave * 16 + 16; ++s) {
    const float* ep = enc_proj + ((size_t)s * B_ + b) * H_;
    float p = 0.0f;
    for (int h = lane; h < H_; h += 32)
      p += tanhf(q[b * H_ + h] + ep[h]) * vvec[h];
#pragma unroll
    for (int off = 16; off > 0; off >>= 1)
      p += __shfl_xor(p, off, 32);
    if (lane == 0) es[s] = p;
  }
  __syncthreads();

  float mx = -3.4e38f;
  for (int s = 0; s < S_; ++s) mx = fmaxf(mx, es[s]);
  float sum = 0.0f;
  for (int s = 0; s < S_; ++s) sum += expf(es[s] - mx);
  const float inv = 1.0f / sum;
  if (threadIdx.x < S_) ws[threadIdx.x] = expf(es[threadIdx.x] - mx) * inv;
  __syncthreads();

  for (int h = threadIdx.x; h < H_; h += blockDim.x) {
    float acc = 0.0f;
    for (int s = 0; s < S_; ++s)
      acc += ws[s] * (float)enc_out[((size_t)s * B_ + b) * H_ + h];
    ctx16[b * H_ + h] = (_Float16)acc;
  }
}

// ---------------------------------------------------------------------------
// Embedding gathers (f32 -> f16) and zero-fill
// ---------------------------------------------------------------------------
__global__ void k_embed_enc(const int* __restrict__ src, const float* __restrict__ emb,
                            _Float16* __restrict__ xe)
{
  const size_t total = (size_t)S_ * B_ * E_;
  for (size_t o = (size_t)blockIdx.x * blockDim.x + threadIdx.x; o < total;
       o += (size_t)gridDim.x * blockDim.x) {
    const int e  = (int)(o % E_);
    const int sb = (int)(o / E_);
    const int b  = sb % B_;
    const int s  = sb / B_;
    const int tok = src[b * S_ + s];
    xe[o] = (_Float16)emb[(size_t)tok * E_ + e];
  }
}

__global__ void k_embed_dec(const int* __restrict__ tgt, const float* __restrict__ emb,
                            _Float16* __restrict__ xb, int t)
{
  const int o = blockIdx.x * blockDim.x + threadIdx.x;
  if (o >= B_ * E_) return;
  const int e = o & (E_ - 1);
  const int b = o >> 9;
  const int tok = tgt[b * T_ + t];
  xb[o] = (_Float16)emb[(size_t)tok * E_ + e];
}

__global__ void k_zero32(unsigned int* __restrict__ p, size_t n)
{
  for (size_t i = (size_t)blockIdx.x * blockDim.x + threadIdx.x; i < n;
       i += (size_t)gridDim.x * blockDim.x)
    p[i] = 0u;
}

// ---------------------------------------------------------------------------
// Host orchestration
// ---------------------------------------------------------------------------
extern "C" void kernel_launch(void* const* d_in, const int* in_sizes, int n_in,
                              void* d_out, int out_size, void* d_ws, size_t ws_size,
                              hipStream_t stream)
{
  (void)in_sizes; (void)n_in; (void)out_size; (void)ws_size;

  const int*   src     = (const int*)  d_in[0];
  const int*   tgt     = (const int*)  d_in[1];
  const float* emb_enc = (const float*)d_in[2];
  const float* enc_Wih = (const float*)d_in[3];
  const float* enc_Whh = (const float*)d_in[4];
  const float* enc_b   = (const float*)d_in[5];
  const float* emb_dec = (const float*)d_in[6];
  const float* dec_Wih = (const float*)d_in[7];
  const float* dec_Whh = (const float*)d_in[8];
  const float* dec_b   = (const float*)d_in[9];
  const float* att_W1  = (const float*)d_in[10];
  const float* att_W2  = (const float*)d_in[11];
  const float* att_v   = (const float*)d_in[12];
  const float* fc_W    = (const float*)d_in[13];
  const float* fc_b    = (const float*)d_in[14];
  float* outp = (float*)d_out;

  char* wp = (char*)d_ws;
  auto carve = [&](size_t bytes) -> char* {
    char* r = wp;
    wp += (bytes + 255) & ~(size_t)255;
    return r;
  };
  _Float16* Wb_enc = (_Float16*)carve((size_t)(E_ + H_)     * G4H * 2); // gate-interleaved
  _Float16* Wb_dec = (_Float16*)carve((size_t)(E_ + 2 * H_) * G4H * 2); // gate-interleaved
  _Float16* Wb_a1  = (_Float16*)carve((size_t)H_ * H_ * 2);
  _Float16* Wb_a2  = (_Float16*)carve((size_t)H_ * H_ * 2);
  _Float16* Wb_fc  = (_Float16*)carve((size_t)H_ * V_ * 2);
  _Float16* x_enc  = (_Float16*)carve((size_t)S_ * B_ * E_ * 2);        // [S][B][E]
  _Float16* eo16   = (_Float16*)carve((size_t)S_ * B_ * H_ * 2);        // [S][B][H]
  float*    eproj  = (float*)   carve((size_t)S_ * B_ * H_ * 4);        // [S][B][H]
  float*    q      = (float*)   carve((size_t)B_ * H_ * 4);
  float*    c      = (float*)   carve((size_t)B_ * H_ * 4);
  _Float16* h16    = (_Float16*)carve((size_t)B_ * H_ * 2);
  _Float16* ctx16  = (_Float16*)carve((size_t)B_ * H_ * 2);
  _Float16* xdec   = (_Float16*)carve((size_t)B_ * E_ * 2);

  const dim3 blk(256);

  // ---- weight prep ----
  k_pack_b<<<2048, blk, 0, stream>>>(enc_Wih, Wb_enc, G4H, E_,      0,               1);
  k_pack_b<<<2048, blk, 0, stream>>>(enc_Whh, Wb_enc, G4H, H_,      E_ / 32,         1);
  k_pack_b<<<2048, blk, 0, stream>>>(dec_Wih, Wb_dec, G4H, E_ + H_, 0,               1);
  k_pack_b<<<2048, blk, 0, stream>>>(dec_Whh, Wb_dec, G4H, H_,      (E_ + H_) / 32,  1);
  k_pack_b<<<2048, blk, 0, stream>>>(att_W1,  Wb_a1,  H_,  H_,      0,               0);
  k_pack_b<<<2048, blk, 0, stream>>>(att_W2,  Wb_a2,  H_,  H_,      0,               0);
  k_pack_b<<<2048, blk, 0, stream>>>(fc_W,    Wb_fc,  V_,  H_,      0,               0);
  k_embed_enc<<<4096, blk, 0, stream>>>(src, emb_enc, x_enc);

  k_zero32<<<64, blk, 0, stream>>>((unsigned int*)c,   (size_t)B_ * H_);
  k_zero32<<<64, blk, 0, stream>>>((unsigned int*)h16, (size_t)B_ * H_ / 2);

  // ---- encoder: 128 sequential fused LSTM steps ----
  for (int s = 0; s < S_; ++s) {
    k_gemm_lstm<<<dim3(32, 1), blk, 0, stream>>>(
        x_enc + (size_t)s * B_ * E_, E_, E_,
        h16, H_, H_,
        (const _Float16*)nullptr, 0, 0,
        Wb_enc, enc_b, c, h16, eo16 + (size_t)s * B_ * H_);
  }

  // ---- enc_proj = enc_out @ att_W2^T  (M = S*B = 8192) ----
  k_gemm<false><<<dim3(H_ / 128, (S_ * B_) / 64), blk, 0, stream>>>(
      eo16, H_, H_,
      (const _Float16*)nullptr, 0, 0,
      (const _Float16*)nullptr, 0, 0,
      Wb_a2, H_ / 16,
      eproj, (long long)H_, (const float*)nullptr);

  // ---- decoder: 127 sequential steps, teacher forcing ----
  for (int t = 0; t < T_ - 1; ++t) {
    k_gemm<false><<<dim3(8, 1), blk, 0, stream>>>(
        h16, H_, H_,
        (const _Float16*)nullptr, 0, 0,
        (const _Float16*)nullptr, 0, 0,
        Wb_a1, H_ / 16,
        q, (long long)H_, (const float*)nullptr);
    k_attn<<<B_, blk, 0, stream>>>(q, eproj, eo16, att_v, ctx16);
    k_embed_dec<<<(B_ * E_) / 256, blk, 0, stream>>>(tgt, emb_dec, xdec, t);
    k_gemm_lstm<<<dim3(32, 1), blk, 0, stream>>>(
        xdec,  E_, E_,
        ctx16, H_, H_,
        h16,   H_, H_,
        Wb_dec, dec_b, c, h16, (_Float16*)nullptr);
    k_gemm<true><<<dim3(63, 1), blk, 0, stream>>>(
        h16, H_, H_,
        (const _Float16*)nullptr, 0, 0,
        (const _Float16*)nullptr, 0, 0,
        Wb_fc, V_ / 16,
        outp + (size_t)t * V_, (long long)(T_ - 1) * V_, fc_b);
  }
}
